// EntityLinker_66142496358618
// MI455X (gfx1250) — matrix operand order repaired and checked by reading, not compiled, not run on hardware
//
#include <hip/hip_runtime.h>
#include <hip/hip_bf16.h>
#include <math.h>

typedef unsigned short u16;
typedef unsigned int   u32;
typedef __attribute__((ext_vector_type(16))) __bf16 v16bf;
typedef __attribute__((ext_vector_type(8)))  float  v8f;
typedef __attribute__((ext_vector_type(4)))  u32    v4u;

#define HDIM 768
#define DDIM 256
#define EDIM 50000
#define MROWS 1024   // B*S
#define EPSV 1e-8f

union Frag { v16bf v; v4u q[2]; };

__device__ __forceinline__ u16 f2bf(float f) {
    u32 u = __float_as_uint(f);
    u32 r = u + 0x7FFFu + ((u >> 16) & 1u);   // round-to-nearest-even
    return (u16)(r >> 16);
}

// ---------------------------------------------------------------------------
// Kernel 1: L2-normalize entity rows, convert to bf16.  One wave per row.
// ---------------------------------------------------------------------------
__global__ __launch_bounds__(256) void ent_norm_kernel(
    const float* __restrict__ ent, u16* __restrict__ out) {
    const int row  = (int)((blockIdx.x * 256u + threadIdx.x) >> 5);
    const int lane = (int)(threadIdx.x & 31u);
    if (row >= EDIM) return;
    const float* r = ent + (size_t)row * DDIM;
    float v[8];
    float s = 0.f;
#pragma unroll
    for (int i = 0; i < 8; ++i) { v[i] = r[lane + i * 32]; s = fmaf(v[i], v[i], s); }
#pragma unroll
    for (int off = 16; off > 0; off >>= 1) s += __shfl_xor(s, off, 32);
    const float scale = 1.0f / fmaxf(sqrtf(s), EPSV);
    u16* o = out + (size_t)row * DDIM;
#pragma unroll
    for (int i = 0; i < 8; ++i) o[lane + i * 32] = f2bf(v[i] * scale);
}

// ---------------------------------------------------------------------------
// Kernel 2: proj = tanh(HS @ W^T + b), row L2-normalize, bf16 out.
// Block = 256 threads (thread t owns column d=t), 8 token rows per block.
// ---------------------------------------------------------------------------
__global__ __launch_bounds__(256) void proj_norm_kernel(
    const float* __restrict__ hs, const float* __restrict__ W,
    const float* __restrict__ bias, u16* __restrict__ out) {
    __shared__ float lhs[8 * HDIM];   // 24 KB
    __shared__ float ssum[8];
    const int t  = (int)threadIdx.x;
    const int m0 = (int)blockIdx.x * 8;
    for (int i = t; i < 8 * HDIM; i += 256) lhs[i] = hs[(size_t)m0 * HDIM + i];
    if (t < 8) ssum[t] = 0.f;
    __syncthreads();

    float acc[8] = {0.f, 0.f, 0.f, 0.f, 0.f, 0.f, 0.f, 0.f};
    const float* wrow = W + (size_t)t * HDIM;
    for (int k = 0; k < HDIM; ++k) {
        const float w = wrow[k];
#pragma unroll
        for (int rr = 0; rr < 8; ++rr) acc[rr] = fmaf(lhs[rr * HDIM + k], w, acc[rr]);
    }
    const float bb = bias[t];
    float v[8];
#pragma unroll
    for (int rr = 0; rr < 8; ++rr) {
        v[rr] = tanhf(acc[rr] + bb);
        atomicAdd(&ssum[rr], v[rr] * v[rr]);
    }
    __syncthreads();
#pragma unroll
    for (int rr = 0; rr < 8; ++rr) {
        const float scale = 1.0f / fmaxf(sqrtf(ssum[rr]), EPSV);
        out[(size_t)(m0 + rr) * DDIM + t] = f2bf(v[rr] * scale);
    }
}

// ---------------------------------------------------------------------------
// Kernel 3: mention-detection head: logits + log_softmax over seq axis.
// One block per batch b (128 threads = seq positions).
// ---------------------------------------------------------------------------
__global__ __launch_bounds__(128) void md_kernel(
    const float* __restrict__ hs, const float* __restrict__ mdW,
    const float* __restrict__ mdb, float* __restrict__ out) {
    __shared__ float lg[128 * 3];
    __shared__ float stat[3];
    const int s = (int)threadIdx.x;
    const int b = (int)blockIdx.x;
    const float* row = hs + ((size_t)b * 128 + s) * HDIM;
    float a0 = 0.f, a1 = 0.f, a2 = 0.f;
    for (int k = 0; k < HDIM; ++k) {
        const float h = row[k];
        a0 = fmaf(h, mdW[k], a0);
        a1 = fmaf(h, mdW[HDIM + k], a1);
        a2 = fmaf(h, mdW[2 * HDIM + k], a2);
    }
    lg[s * 3 + 0] = a0 + mdb[0];
    lg[s * 3 + 1] = a1 + mdb[1];
    lg[s * 3 + 2] = a2 + mdb[2];
    __syncthreads();
    if (s < 3) {
        float mx = -INFINITY;
        for (int i = 0; i < 128; ++i) mx = fmaxf(mx, lg[i * 3 + s]);
        float se = 0.f;
        for (int i = 0; i < 128; ++i) se += expf(lg[i * 3 + s] - mx);
        stat[s] = mx + logf(se);
    }
    __syncthreads();
#pragma unroll
    for (int c = 0; c < 3; ++c)
        out[((size_t)b * 128 + s) * 3 + c] = lg[s * 3 + c] - stat[c];
}

// ---------------------------------------------------------------------------
// Kernel 4: similarity GEMM  C[1024,50000] = A(bf16)[1024,256] @ B(bf16)^T.
// One wave per 32x80 output tile: MT=2 x NT=5 tiles of 16x16, K in 8 steps
// of 32, v_wmma_f32_16x16x32_bf16.
// Fragment layouts per CDNA5 ISA 7.12.2:
//   A 16x32 bf16: lane<16 -> M=lane, v0..3=K 0..7, v4..7=K 16..23;
//                 lane>=16 -> v0..3=K 8..15, v4..7=K 24..31.
//   B 32x16 bf16: lane<16 -> N=lane, v0..3=K 0..7, v4..7=K 16..23; lane>=16 -> K 8..15, 24..31.
//   C 16x16 f32 : vgpr r -> M = r + 8*(lane>=16), N = lane%16.
// ---------------------------------------------------------------------------
__global__ __launch_bounds__(32) void sim_gemm_kernel(
    const u16* __restrict__ A, const u16* __restrict__ Bm,
    float* __restrict__ C) {
    const int lane = (int)(threadIdx.x & 31u);
    const int half = lane >> 4;
    const int lq   = lane & 15;
    const int m0 = (int)blockIdx.x * 32;   // 2 M-tiles
    const int n0 = (int)blockIdx.y * 80;   // 5 N-tiles

    v8f acc[2][5] = {};

    for (int ks = 0; ks < 8; ++ks) {
        const int k0 = ks * 32;
        Frag a[2];
#pragma unroll
        for (int r = 0; r < 2; ++r) {
            const u16* ap = A + (size_t)(m0 + r * 16 + lq) * DDIM + k0 + half * 8;
            a[r].q[0] = *(const v4u*)(ap);        // K k0+half*8 .. +7
            a[r].q[1] = *(const v4u*)(ap + 16);   // K k0+16+half*8 .. +7
        }
#pragma unroll
        for (int t = 0; t < 5; ++t) {
            Frag bf;
            const u16* bp = Bm + (size_t)(n0 + t * 16 + lq) * DDIM + k0 + half * 8;
            bf.q[0] = *(const v4u*)(bp);          // K k0+half*8 .. +7
            bf.q[1] = *(const v4u*)(bp + 16);     // K k0+16+half*8 .. +7
            acc[0][t] = __builtin_amdgcn_wmma_f32_16x16x32_bf16(
                false, a[0].v, false, bf.v, (short)0, acc[0][t], false, false);
            acc[1][t] = __builtin_amdgcn_wmma_f32_16x16x32_bf16(
                false, a[1].v, false, bf.v, (short)0, acc[1][t], false, false);
        }
    }

#pragma unroll
    for (int r = 0; r < 2; ++r)
#pragma unroll
        for (int t = 0; t < 5; ++t)
#pragma unroll
            for (int i = 0; i < 8; ++i) {
                const int m = m0 + r * 16 + i + half * 8;
                const int n = n0 + t * 16 + lq;
                C[(size_t)m * EDIM + n] = acc[r][t][i];
            }
}

// ---------------------------------------------------------------------------
extern "C" void kernel_launch(void* const* d_in, const int* in_sizes, int n_in,
                              void* d_out, int out_size, void* d_ws, size_t ws_size,
                              hipStream_t stream) {
    const float* hs   = (const float*)d_in[0];  // [8,128,768]
    const float* W    = (const float*)d_in[1];  // [256,768]
    const float* bias = (const float*)d_in[2];  // [256]
    const float* mdW  = (const float*)d_in[3];  // [3,768]
    const float* mdb  = (const float*)d_in[4];  // [3]
    const float* ent  = (const float*)d_in[5];  // [50000,256]

    float* md_out = (float*)d_out;              // 8*128*3 = 3072
    float* sim    = (float*)d_out + 3072;       // 1024*50000

    u16* entbf  = (u16*)d_ws;                                   // 25,600,000 B
    u16* projbf = (u16*)((char*)d_ws + (size_t)EDIM * DDIM * 2); // 524,288 B

    // entity table normalize + bf16 (one wave per row; 8 rows per block)
    ent_norm_kernel<<<EDIM / 8, 256, 0, stream>>>(ent, entbf);
    // projection head: tanh + normalize + bf16 (8 token rows per block)
    proj_norm_kernel<<<MROWS / 8, 256, 0, stream>>>(hs, W, bias, projbf);
    // mention-detection head
    md_kernel<<<8, 128, 0, stream>>>(hs, mdW, mdb, md_out);
    // big bf16 WMMA GEMM: grid 32 (M) x 625 (N)
    sim_gemm_kernel<<<dim3(32, 625), 32, 0, stream>>>(projbf, entbf, sim);
}
